// nanoGPT_61632780697830
// MI455X (gfx1250) — compile-verified, matrix operand order checked by reading, not run on hardware
//
#include <hip/hip_runtime.h>
#include <hip/hip_bf16.h>
#include <math.h>

// ---------------- model constants ----------------
#define GV   50257
#define GVP  50304      // V padded to multiple of 64 for aligned bf16 weight rows
#define GC   768
#define GH   12
#define GL   12
#define GB   4
#define GT   1024
#define GBT  (GB*GT)    // 4096
#define GC3  (3*GC)     // 2304
#define GC4  (4*GC)     // 3072

typedef __attribute__((ext_vector_type(16))) __bf16 v16bf;
typedef __attribute__((ext_vector_type(8)))  __bf16 v8bf;   // 16 bytes
typedef __attribute__((ext_vector_type(8)))  float  v8f;

#define WMMA_BF16(a, b, c) \
  __builtin_amdgcn_wmma_f32_16x16x32_bf16(false, (a), false, (b), (short)0, (c), false, false)

// ---------------- f32 -> bf16 conversion (with optional column padding) ----
__global__ __launch_bounds__(256)
void convert_bf16_pad(const float* __restrict__ src, __bf16* __restrict__ dst,
                      int rows, int cols, int ldd) {
  size_t i = (size_t)blockIdx.x * 256 + threadIdx.x;
  size_t total = (size_t)rows * ldd;
  if (i >= total) return;
  int c = (int)(i % ldd);
  size_t r = i / ldd;
  float v = (c < cols) ? src[r * (size_t)cols + c] : 0.0f;
  dst[i] = (__bf16)v;
}

// ---------------- token + position embedding ------------------------------
__global__ __launch_bounds__(256)
void embed_kernel(const int* __restrict__ idx, const float* __restrict__ wte,
                  const float* __restrict__ wpe, float* __restrict__ x) {
  size_t i = (size_t)blockIdx.x * 256 + threadIdx.x;
  if (i >= (size_t)GBT * GC) return;
  int c = (int)(i % GC);
  size_t bt = i / GC;
  int t = (int)(bt % GT);
  x[i] = wte[(size_t)idx[bt] * GC + c] + wpe[(size_t)t * GC + c];
}

// ---------------- layernorm: f32 in -> bf16 out ---------------------------
__global__ __launch_bounds__(256)
void layernorm_kernel(const float* __restrict__ x, const float* __restrict__ w,
                      const float* __restrict__ b, __bf16* __restrict__ out) {
  __shared__ float red[256];
  int row = blockIdx.x;
  const float* xr = x + (size_t)row * GC;
  int tid = threadIdx.x;

  float s = 0.f;
  for (int i = tid; i < GC; i += 256) s += xr[i];
  red[tid] = s; __syncthreads();
  for (int o = 128; o > 0; o >>= 1) { if (tid < o) red[tid] += red[tid + o]; __syncthreads(); }
  float mu = red[0] / GC;
  __syncthreads();

  float vsum = 0.f;
  for (int i = tid; i < GC; i += 256) { float d = xr[i] - mu; vsum += d * d; }
  red[tid] = vsum; __syncthreads();
  for (int o = 128; o > 0; o >>= 1) { if (tid < o) red[tid] += red[tid + o]; __syncthreads(); }
  float rstd = rsqrtf(red[0] / GC + 1e-5f);

  for (int i = tid; i < GC; i += 256)
    out[(size_t)row * GC + i] = (__bf16)((xr[i] - mu) * rstd * w[i] + b[i]);
}

// ---------------- WMMA GEMM ------------------------------------------------
// C = A(bf16,[M,K]) x B(bf16,[K,N]) + bias [+gelu] [+residual]
// Block tile 128x64, BK=32. 8 waves in 4x2 grid; each wave owns a 32x32
// output tile -> 4 v_wmma per K-step.  Staging is UNCONDITIONAL:
// M %128==0, K %32==0, Nload %64==0, B valid out to Nload columns.
#define LSTR 40   // LDS row stride in elements (80B, 16B-aligned rows)

__device__ __forceinline__ float gelu_tanh(float x) {
  float x3 = x * x * x;
  return 0.5f * x * (1.0f + tanhf(0.7978845608028654f * (x + 0.044715f * x3)));
}

__global__ __launch_bounds__(256)
void gemm_bf16_wmma(const __bf16* __restrict__ A, int lda,
                    const __bf16* __restrict__ Bm, int ldb,
                    const float* __restrict__ bias,
                    const float* __restrict__ residual, int ldres,
                    float* __restrict__ outF, __bf16* __restrict__ outB, int ldc,
                    int M, int N, int K, int gelu_flag) {
  __shared__ __bf16 As[128 * LSTR];
  __shared__ __bf16 Bt[64 * LSTR];    // B tile stored transposed: Bt[n][k]

  int tid  = threadIdx.x;
  int lane = tid & 31;
  int wave = tid >> 5;
  int blockM = blockIdx.y * 128;
  int blockN = blockIdx.x * 64;
  int wm = (wave >> 1) * 32;
  int wn = (wave & 1) * 32;

  v8f acc00 = {}, acc01 = {}, acc10 = {}, acc11 = {};

  int ar = tid >> 1;                  // 0..127
  int ac = (tid & 1) * 16;            // 0 or 16
  int br = tid >> 3;                  // 0..31 (k within tile)
  int bc = (tid & 7) * 8;             // 0..56 (n within tile)

  const __bf16* ag = A  + (size_t)(blockM + ar) * lda + ac;
  const __bf16* bg = Bm + (size_t)br * ldb + (blockN + bc);

  for (int k0 = 0; k0 < K; k0 += 32) {
    {
      v8bf a0 = *(const v8bf*)(ag + k0);
      v8bf a1 = *(const v8bf*)(ag + k0 + 8);
      *(v8bf*)&As[ar * LSTR + ac]     = a0;
      *(v8bf*)&As[ar * LSTR + ac + 8] = a1;
    }
    {
      v8bf bvec = *(const v8bf*)(bg + (size_t)k0 * ldb);
      #pragma unroll
      for (int i = 0; i < 8; i++) Bt[(bc + i) * LSTR + br] = bvec[i];
    }
    if (k0 + 32 < K) {
      __builtin_prefetch(ag + k0 + 32, 0, 1);
      __builtin_prefetch(bg + (size_t)(k0 + 32) * ldb, 0, 1);
    }
    __syncthreads();

    int rsel = lane & 15;
    int koff = (lane >> 4) * 8;
    const __bf16* ap0 = &As[(wm + rsel) * LSTR + koff];
    const __bf16* ap1 = &As[(wm + 16 + rsel) * LSTR + koff];
    const __bf16* bp0 = &Bt[(wn + rsel) * LSTR + koff];
    const __bf16* bp1 = &Bt[(wn + 16 + rsel) * LSTR + koff];
    v16bf a0, a1, b0, b1;
    #pragma unroll
    for (int i = 0; i < 8; i++) {
      a0[i] = ap0[i];  a0[i + 8] = ap0[16 + i];
      a1[i] = ap1[i];  a1[i + 8] = ap1[16 + i];
      b0[i] = bp0[i];  b0[i + 8] = bp0[16 + i];
      b1[i] = bp1[i];  b1[i + 8] = bp1[16 + i];
    }
    acc00 = WMMA_BF16(a0, b0, acc00);
    acc01 = WMMA_BF16(a0, b1, acc01);
    acc10 = WMMA_BF16(a1, b0, acc10);
    acc11 = WMMA_BF16(a1, b1, acc11);
    __syncthreads();
  }

  int cl = lane & 15;
  int rh = (lane >> 4) * 8;
  #pragma unroll
  for (int si = 0; si < 2; si++) {
    #pragma unroll
    for (int sj = 0; sj < 2; sj++) {
      v8f acc = (si == 0) ? ((sj == 0) ? acc00 : acc01)
                          : ((sj == 0) ? acc10 : acc11);
      int gn = blockN + wn + sj * 16 + cl;
      if (gn >= N) continue;
      float bv = bias ? bias[gn] : 0.0f;
      #pragma unroll
      for (int r = 0; r < 8; r++) {
        int gm = blockM + wm + si * 16 + rh + r;
        float val = acc[r] + bv;
        if (gelu_flag) val = gelu_tanh(val);
        if (residual) val += residual[(size_t)gm * ldres + gn];
        if (outF) outF[(size_t)gm * ldc + gn] = val;
        if (outB) outB[(size_t)gm * ldc + gn] = (__bf16)val;
      }
    }
  }
}

// ---------------- V transpose: qkvbf[t][2C + h*64 + d] -> vt[bh][d][t] -----
__global__ __launch_bounds__(256)
void transpose_v_kernel(const __bf16* __restrict__ qkvbf, __bf16* __restrict__ vt) {
  size_t i = (size_t)blockIdx.x * 256 + threadIdx.x;
  if (i >= (size_t)GB * GH * 64 * GT) return;
  int t  = (int)(i % GT);
  size_t rem = i / GT;
  int d  = (int)(rem % 64);
  int bh = (int)(rem / 64);
  int b  = bh / GH;
  int h  = bh % GH;
  vt[i] = qkvbf[(size_t)(b * GT + t) * GC3 + 2 * GC + h * 64 + d];
}

// ---------------- flash attention with WMMA --------------------------------
// One wave32 per 16-query tile. Per 32-key block:
//   S(16x32) = Q(16x64) K^T  -> 4 wmma (2 key subtiles x 2 dim chunks)
//   online softmax in registers (row stats per lane-half, shfl_xor reductions)
//   P relayout via wave-private LDS tile (C-layout -> A-layout)
//   O(16x64) += P V          -> 4 wmma (4 dim subtiles)
// Q/K read straight from bf16 qkv (B-layout wants K[key][dim] contiguous);
// V read from transposed vt[dim][key].
__global__ __launch_bounds__(256)
void attention_wmma_kernel(const __bf16* __restrict__ qkvbf,
                           const __bf16* __restrict__ vt,
                           __bf16* __restrict__ y) {
  __shared__ __bf16 plbuf[8 * 16 * LSTR];

  int lane = threadIdx.x & 31;
  int wv   = threadIdx.x >> 5;
  int gw   = blockIdx.x * 8 + wv;       // bh * (T/16) + qtile
  int qt   = gw % (GT / 16);
  int bh   = gw / (GT / 16);
  int b    = bh / GH;
  int h    = bh % GH;
  int q0   = qt * 16;

  int rsel = lane & 15;                 // fragment row / column select
  int koff = (lane >> 4) * 8;           // K-offset split (also C-layout row half)
  int cl   = rsel;
  int rh   = koff;

  __bf16* Pl = &plbuf[wv * 16 * LSTR];

  // Q A-fragments: dims 0..31 (aq0) and 32..63 (aq1); row = query q0+rsel
  const __bf16* qrow = qkvbf + (size_t)(b * GT + q0 + rsel) * GC3 + h * 64;
  v16bf aq0, aq1;
  #pragma unroll
  for (int i = 0; i < 8; i++) {
    aq0[i] = qrow[koff + i];       aq0[i + 8] = qrow[koff + 16 + i];
    aq1[i] = qrow[32 + koff + i];  aq1[i + 8] = qrow[32 + koff + 16 + i];
  }

  float mrow[8], lrow[8];
  #pragma unroll
  for (int r = 0; r < 8; r++) { mrow[r] = -3.0e38f; lrow[r] = 0.f; }
  v8f o0 = {}, o1 = {}, o2 = {}, o3 = {};

  const __bf16* kbase = qkvbf + (size_t)(b * GT) * GC3 + GC + h * 64;
  const __bf16* vtb   = vt + (size_t)bh * 64 * GT;

  for (int kb = 0; kb <= q0 + 15; kb += 32) {
    // ---- S = Q K^T ----
    v8f s0 = {}, s1 = {};
    {
      const __bf16* krow = kbase + (size_t)(kb + rsel) * GC3;
      v16bf bk0, bk1;
      #pragma unroll
      for (int i = 0; i < 8; i++) {
        bk0[i] = krow[koff + i];       bk0[i + 8] = krow[koff + 16 + i];
        bk1[i] = krow[32 + koff + i];  bk1[i + 8] = krow[32 + koff + 16 + i];
      }
      s0 = WMMA_BF16(aq0, bk0, s0);
      s0 = WMMA_BF16(aq1, bk1, s0);
    }
    {
      const __bf16* krow = kbase + (size_t)(kb + 16 + rsel) * GC3;
      v16bf bk0, bk1;
      #pragma unroll
      for (int i = 0; i < 8; i++) {
        bk0[i] = krow[koff + i];       bk0[i + 8] = krow[koff + 16 + i];
        bk1[i] = krow[32 + koff + i];  bk1[i + 8] = krow[32 + koff + 16 + i];
      }
      s1 = WMMA_BF16(aq0, bk0, s1);
      s1 = WMMA_BF16(aq1, bk1, s1);
    }

    // ---- scale + causal mask (C-layout: row=q0+rh+r, key=kb+sj*16+cl) ----
    #pragma unroll
    for (int r = 0; r < 8; r++) {
      int row = q0 + rh + r;
      float v0 = s0[r] * 0.125f;
      float v1 = s1[r] * 0.125f;
      if (kb + cl      > row) v0 = -3.0e38f;
      if (kb + 16 + cl > row) v1 = -3.0e38f;
      s0[r] = v0; s1[r] = v1;
    }

    // ---- block row max ----
    float bm[8];
    #pragma unroll
    for (int r = 0; r < 8; r++) bm[r] = fmaxf(s0[r], s1[r]);
    #pragma unroll
    for (int o = 8; o > 0; o >>= 1) {
      #pragma unroll
      for (int r = 0; r < 8; r++) bm[r] = fmaxf(bm[r], __shfl_xor(bm[r], o, 32));
    }

    // ---- online softmax update ----
    float scl[8];
    #pragma unroll
    for (int r = 0; r < 8; r++) {
      float mn = fmaxf(mrow[r], bm[r]);
      scl[r] = __expf(mrow[r] - mn);
      mrow[r] = mn;
    }
    float rs[8];
    #pragma unroll
    for (int r = 0; r < 8; r++) {
      float p0 = __expf(s0[r] - mrow[r]);
      float p1 = __expf(s1[r] - mrow[r]);
      s0[r] = p0; s1[r] = p1;
      rs[r] = p0 + p1;
    }
    #pragma unroll
    for (int o = 8; o > 0; o >>= 1) {
      #pragma unroll
      for (int r = 0; r < 8; r++) rs[r] += __shfl_xor(rs[r], o, 32);
    }
    #pragma unroll
    for (int r = 0; r < 8; r++) {
      lrow[r] = lrow[r] * scl[r] + rs[r];
      o0[r] *= scl[r]; o1[r] *= scl[r]; o2[r] *= scl[r]; o3[r] *= scl[r];
    }

    // ---- P: C-layout -> LDS [row][key] -> A-fragment ----
    #pragma unroll
    for (int r = 0; r < 8; r++) {
      Pl[(rh + r) * LSTR + cl]      = (__bf16)s0[r];
      Pl[(rh + r) * LSTR + 16 + cl] = (__bf16)s1[r];
    }
    // same-wave LDS ops are in-order on CDNA5 -> no barrier needed
    const __bf16* pp = &Pl[rsel * LSTR + koff];
    v16bf ap;
    #pragma unroll
    for (int i = 0; i < 8; i++) { ap[i] = pp[i]; ap[i + 8] = pp[16 + i]; }

    // ---- O += P V ----
    #pragma unroll
    for (int dt = 0; dt < 4; dt++) {
      const __bf16* vr = vtb + (size_t)(dt * 16 + rsel) * GT + kb + koff;
      v16bf bv;
      #pragma unroll
      for (int i = 0; i < 8; i++) { bv[i] = vr[i]; bv[i + 8] = vr[16 + i]; }
      if      (dt == 0) o0 = WMMA_BF16(ap, bv, o0);
      else if (dt == 1) o1 = WMMA_BF16(ap, bv, o1);
      else if (dt == 2) o2 = WMMA_BF16(ap, bv, o2);
      else              o3 = WMMA_BF16(ap, bv, o3);
    }
  }

  // ---- epilogue: y[row][h*64 + dt*16 + cl] = O / l ----
  #pragma unroll
  for (int r = 0; r < 8; r++) {
    float inv = 1.0f / lrow[r];
    __bf16* yr = y + (size_t)(b * GT + q0 + rh + r) * GC + h * 64 + cl;
    yr[0]  = (__bf16)(o0[r] * inv);
    yr[16] = (__bf16)(o1[r] * inv);
    yr[32] = (__bf16)(o2[r] * inv);
    yr[48] = (__bf16)(o3[r] * inv);
  }
}

// ---------------- cross-entropy loss over V logits -------------------------
__global__ void loss_init(float* accum) { accum[0] = 0.f; accum[1] = 0.f; }

__global__ __launch_bounds__(256)
void loss_kernel(const float* __restrict__ logits, const int* __restrict__ targets,
                 float* accum) {
  __shared__ float red[256];
  int row = blockIdx.x;
  int tid = threadIdx.x;
  const float* lr = logits + (size_t)row * GV;

  float m = -3.0e38f;
  for (int i = tid; i < GV; i += 256) m = fmaxf(m, lr[i]);
  red[tid] = m; __syncthreads();
  for (int o = 128; o > 0; o >>= 1) { if (tid < o) red[tid] = fmaxf(red[tid], red[tid + o]); __syncthreads(); }
  m = red[0]; __syncthreads();

  float s = 0.f;
  for (int i = tid; i < GV; i += 256) s += __expf(lr[i] - m);
  red[tid] = s; __syncthreads();
  for (int o = 128; o > 0; o >>= 1) { if (tid < o) red[tid] += red[tid + o]; __syncthreads(); }

  if (tid == 0) {
    int t = targets[row];
    float valid = (t != -1) ? 1.0f : 0.0f;
    int tc = t < 0 ? 0 : (t > GV - 1 ? GV - 1 : t);
    float nll = (logf(red[0]) + m) - lr[tc];
    atomicAdd(&accum[0], nll * valid);
    atomicAdd(&accum[1], valid);
  }
}

__global__ void loss_final(const float* accum, float* out) {
  out[0] = accum[0] / fmaxf(accum[1], 1.0f);
}

// ---------------- orchestration --------------------------------------------
extern "C" void kernel_launch(void* const* d_in, const int* in_sizes, int n_in,
                              void* d_out, int out_size, void* d_ws, size_t ws_size,
                              hipStream_t stream) {
  const int*   idx     = (const int*)  d_in[0];
  const int*   targets = (const int*)  d_in[1];
  const float* wte     = (const float*)d_in[2];
  const float* wpe     = (const float*)d_in[3];
  const float* ln1_w   = (const float*)d_in[4];
  const float* ln1_b   = (const float*)d_in[5];
  const float* qkv_w   = (const float*)d_in[6];
  const float* qkv_b   = (const float*)d_in[7];
  const float* proj_w  = (const float*)d_in[8];
  const float* proj_b  = (const float*)d_in[9];
  const float* ln2_w   = (const float*)d_in[10];
  const float* ln2_b   = (const float*)d_in[11];
  const float* fc_w    = (const float*)d_in[12];
  const float* fc_b    = (const float*)d_in[13];
  const float* fc2_w   = (const float*)d_in[14];
  const float* fc2_b   = (const float*)d_in[15];
  const float* lnf_w   = (const float*)d_in[16];
  const float* lnf_b   = (const float*)d_in[17];
  const float* head_w  = (const float*)d_in[18];

  // workspace carve-up (256B aligned)
  char* ws = (char*)d_ws;
  size_t off = 0;
  auto carve = [&](size_t bytes) -> void* {
    void* p = ws + off;
    off += (bytes + 255) & ~(size_t)255;
    return p;
  };
  float*  x     = (float*) carve((size_t)GBT * GC  * 4);
  __bf16* hb    = (__bf16*)carve((size_t)GBT * GC  * 2);
  __bf16* qkvbf = (__bf16*)carve((size_t)GBT * GC3 * 2);
  __bf16* vt    = (__bf16*)carve((size_t)GB * GH * 64 * GT * 2);
  __bf16* yb    = (__bf16*)carve((size_t)GBT * GC  * 2);
  __bf16* gb    = (__bf16*)carve((size_t)GBT * GC4 * 2);
  __bf16* wqkv  = (__bf16*)carve((size_t)GL * GC  * GC3 * 2);
  __bf16* wproj = (__bf16*)carve((size_t)GL * GC  * GC  * 2);
  __bf16* wfc   = (__bf16*)carve((size_t)GL * GC  * GC4 * 2);
  __bf16* wfc2  = (__bf16*)carve((size_t)GL * GC4 * GC  * 2);
  __bf16* whead = (__bf16*)carve((size_t)GC * GVP * 2);
  float*  accum = (float*) carve(2 * 4);

  auto cvt = [&](const float* src, __bf16* dst, size_t rows, int cols, int ldd) {
    size_t total = rows * (size_t)ldd;
    convert_bf16_pad<<<dim3((unsigned)((total + 255) / 256)), dim3(256), 0, stream>>>(
        src, dst, (int)rows, cols, ldd);
  };
  cvt(qkv_w,  wqkv,  (size_t)GL * GC,  GC3, GC3);
  cvt(proj_w, wproj, (size_t)GL * GC,  GC,  GC);
  cvt(fc_w,   wfc,   (size_t)GL * GC,  GC4, GC4);
  cvt(fc2_w,  wfc2,  (size_t)GL * GC4, GC,  GC);
  cvt(head_w, whead, (size_t)GC,       GV,  GVP);

  // embedding
  {
    size_t total = (size_t)GBT * GC;
    embed_kernel<<<dim3((unsigned)((total + 255) / 256)), dim3(256), 0, stream>>>(
        idx, wte, wpe, x);
  }

  auto gemm = [&](const __bf16* A, int lda, const __bf16* Bm, int ldb,
                  const float* bias, const float* residual, int ldres,
                  float* outF, __bf16* outB, int ldc,
                  int M, int N, int Nload, int K, int gelu_flag) {
    dim3 grid((unsigned)(Nload / 64), (unsigned)(M / 128));
    gemm_bf16_wmma<<<grid, dim3(256), 0, stream>>>(
        A, lda, Bm, ldb, bias, residual, ldres, outF, outB, ldc, M, N, K, gelu_flag);
  };

  for (int l = 0; l < GL; l++) {
    // ln1 -> hb
    layernorm_kernel<<<GBT, 256, 0, stream>>>(x, ln1_w + l * GC, ln1_b + l * GC, hb);
    // qkv (bf16 out) = hb @ Wqkv + b     [4096 x 2304]
    gemm(hb, GC, wqkv + (size_t)l * GC * GC3, GC3,
         qkv_b + (size_t)l * GC3, nullptr, 0,
         nullptr, qkvbf, GC3, GBT, GC3, GC3, GC, 0);
    // vt[bh][d][t] transpose of V slice
    {
      size_t total = (size_t)GB * GH * 64 * GT;
      transpose_v_kernel<<<dim3((unsigned)((total + 255) / 256)), dim3(256), 0, stream>>>(
          qkvbf, vt);
    }
    // flash attention (WMMA) -> yb (bf16)
    attention_wmma_kernel<<<(GB * GH * (GT / 16)) / 8, 256, 0, stream>>>(qkvbf, vt, yb);
    // x = x + yb @ Wproj + b
    gemm(yb, GC, wproj + (size_t)l * GC * GC, GC,
         proj_b + (size_t)l * GC, x, GC,
         x, nullptr, GC, GBT, GC, GC, GC, 0);
    // ln2 -> hb
    layernorm_kernel<<<GBT, 256, 0, stream>>>(x, ln2_w + l * GC, ln2_b + l * GC, hb);
    // gb = gelu(hb @ Wfc + b)   (bf16 out)
    gemm(hb, GC, wfc + (size_t)l * GC * GC4, GC4,
         fc_b + (size_t)l * GC4, nullptr, 0,
         nullptr, gb, GC4, GBT, GC4, GC4, GC, 1);
    // x = x + gb @ Wfc2 + b
    gemm(gb, GC4, wfc2 + (size_t)l * GC4 * GC, GC,
         fc2_b + (size_t)l * GC, x, GC,
         x, nullptr, GC, GBT, GC, GC, GC4, 0);
  }

  // final LN + head
  layernorm_kernel<<<GBT, 256, 0, stream>>>(x, lnf_w, lnf_b, hb);
  float* logits = (float*)d_out;
  gemm(hb, GC, whead, GVP, nullptr, nullptr, 0,
       logits, nullptr, GV, GBT, GV, GVP, GC, 0);

  // loss
  loss_init<<<1, 1, 0, stream>>>(accum);
  loss_kernel<<<GBT, 256, 0, stream>>>(logits, targets, accum);
  loss_final<<<1, 1, 0, stream>>>(accum, logits + (size_t)GBT * GV);
}
